// MHA_42116449304883
// MI455X (gfx1250) — matrix operand order repaired and checked by reading, not compiled, not run on hardware
//
#include <hip/hip_runtime.h>
#include <hip/hip_bf16.h>

typedef __attribute__((ext_vector_type(16))) _Float16 v16h;
typedef __attribute__((ext_vector_type(8)))  _Float16 h8;
typedef __attribute__((ext_vector_type(4)))  _Float16 h4;
typedef __attribute__((ext_vector_type(4)))  float    f4;
typedef __attribute__((ext_vector_type(8)))  float    v8f;

#define SEQ   2048
#define BATCH 4
#define DIM   512
#define HEADS 8
#define DH    64

__device__ __forceinline__ v8f wmma_f16(v16h a, v16h b, v8f c) {
  return __builtin_amdgcn_wmma_f32_16x16x32_f16(
      false, a, false, b, (short)0, c, false, false);
}

// A-fragment (16x32 f16) per lane from row-major f16 matrix, row stride ldm
// (halves). m = lane%16; two contiguous 8-half chunks at k = kb and kb+8.
__device__ __forceinline__ v16h load_afrag(const _Float16* base, int ldm,
                                           int lmod, int lhalf) {
  int kb = lhalf ? 16 : 0;
  h8 lo = *(const h8*)(base + (size_t)lmod * ldm + kb);
  h8 hi = *(const h8*)(base + (size_t)lmod * ldm + kb + 8);
  v16h a;
#pragma unroll
  for (int e = 0; e < 8; e++) { a[e] = lo[e]; a[e + 8] = hi[e]; }
  return a;
}

// ---------------------------------------------------------------------------
// Kernel 0: one-time fp32 -> f16 conversion of the 4 weight matrices.
// grid (256, 4), 256 thr: each thread converts exactly 4 elements.
// ---------------------------------------------------------------------------
__global__ __launch_bounds__(256) void wcvt_kernel(
    const float* __restrict__ Wq, const float* __restrict__ Wk,
    const float* __restrict__ Wv, const float* __restrict__ Wo,
    _Float16* __restrict__ W16) {  // 4 matrices back-to-back, 262144 halves ea.
  const float* srcs[4] = {Wq, Wk, Wv, Wo};
  const float* s = srcs[blockIdx.y];
  _Float16* d = W16 + (size_t)blockIdx.y * DIM * DIM;
  size_t i = ((size_t)blockIdx.x * 256 + threadIdx.x) * 4;
  f4 v = *(const f4*)(s + i);
  h4 o;
#pragma unroll
  for (int e = 0; e < 4; e++) o[e] = (_Float16)v[e];
  *(h4*)(d + i) = o;
}

// ---------------------------------------------------------------------------
// Kernel 1: QKV projection + RoPE.  Q gets the 1/sqrt(dh) scale folded in.
// Q,K,V stored f16 as [b][h][n][dh].
// ---------------------------------------------------------------------------
__global__ __launch_bounds__(256) void qkv_rope_kernel(
    const float* __restrict__ x, const float* __restrict__ rope,
    const _Float16* __restrict__ W16,   // [Wq16 | Wk16 | Wv16 | Wo16]
    _Float16* __restrict__ Q, _Float16* __restrict__ K,
    _Float16* __restrict__ V) {
  __shared__ __align__(32) _Float16 lx[16][DIM];
  const int n0 = blockIdx.x * 16;
  const int bi = blockIdx.y;
  const int tid = threadIdx.x;

  for (int i = tid; i < 16 * DIM; i += 256) {
    int r = i >> 9, c = i & 511;
    lx[r][c] = (_Float16)x[((size_t)(n0 + r) * BATCH + bi) * DIM + c];
  }
  __syncthreads();

  const int wave = tid >> 5, lane = tid & 31;
  const int lhalf = lane >> 4, lmod = lane & 15;
  _Float16* Outs[3] = {Q, K, V};

  for (int t = wave; t < 96; t += 8) {       // 3 mats * 32 col tiles
    const int mat = t >> 5;
    const int oc0 = (t & 31) << 4;
    const _Float16* W = W16 + (size_t)mat * DIM * DIM;
    v8f c = {};
#pragma unroll 4
    for (int kk = 0; kk < DIM; kk += 32) {
      v16h a = load_afrag(&lx[0][kk], DIM, lmod, lhalf);
      // B fragment: B[k][o] = W[o][k]; 16 contiguous f16 per lane (32B).
      v16h bf = *(const v16h*)(W + (size_t)(oc0 + lmod) * DIM + kk + lhalf * 16);
      c = wmma_f16(a, bf, c);
    }
    const int o = oc0 + lmod;                 // output channel
    const int h = o >> 6, d = o & 63;
    if (mat == 2) {                           // V: store directly
#pragma unroll
      for (int v = 0; v < 8; v++) {
        int m = lhalf * 8 + v;
        size_t idx = (((size_t)bi * HEADS + h) * SEQ + (n0 + m)) * DH + d;
        Outs[2][idx] = (_Float16)c[v];
      }
    } else {                                  // Q/K: RoPE (+scale for Q)
      const float qs = (mat == 0) ? 0.125f : 1.0f;  // dh^-0.5 = 1/8
#pragma unroll
      for (int v = 0; v < 8; v++) {
        int m = lhalf * 8 + v;
        float partner = __shfl_xor(c[v], 1, 32);    // value at channel o^1
        float rot = (o & 1) ? partner : -partner;   // rotate_half
        float f = rope[((size_t)bi * SEQ + (n0 + m)) * DH + d];
        // rope values are uniform [0,1): hardware trans ops are exact enough
        float val = (c[v] * __cosf(f) + rot * __sinf(f)) * qs;
        size_t idx = (((size_t)bi * HEADS + h) * SEQ + (n0 + m)) * DH + d;
        Outs[mat][idx] = (_Float16)val;
      }
    }
  }
}

// ---------------------------------------------------------------------------
// Kernel 2: causal flash attention per (b,h). 128 query rows per block,
// 16 rows per wave, 32-key tiles. Output * head_scale, stored f16 [b][n][512].
// ---------------------------------------------------------------------------
__global__ __launch_bounds__(256) void attn_kernel(
    const _Float16* __restrict__ Q, const _Float16* __restrict__ K,
    const _Float16* __restrict__ V, const float* __restrict__ head_scale,
    _Float16* __restrict__ AO) {
  __shared__ __align__(32) _Float16 qtile[128][DH];
  __shared__ __align__(32) _Float16 ktile[32][DH];
  __shared__ __align__(32) _Float16 vtile[DH][32];   // transposed V
  __shared__ __align__(32) _Float16 pbuf[8][16][32]; // per-wave P staging

  const int q0 = blockIdx.x * 128;
  const int bh = blockIdx.y;
  const int h = bh & 7, bi = bh >> 3;
  const _Float16* Qp = Q + (size_t)bh * SEQ * DH;
  const _Float16* Kp = K + (size_t)bh * SEQ * DH;
  const _Float16* Vp = V + (size_t)bh * SEQ * DH;
  const int tid = threadIdx.x, wave = tid >> 5, lane = tid & 31;
  const int lhalf = lane >> 4, lmod = lane & 15;

  for (int i = tid * 8; i < 128 * DH; i += 2048)
    *(h8*)(&qtile[0][0] + i) = *(const h8*)(Qp + (size_t)q0 * DH + i);
  __syncthreads();

  const int qrow = q0 + wave * 16;
  v16h aq0 = load_afrag(&qtile[wave * 16][0], DH, lmod, lhalf);
  v16h aq1 = load_afrag(&qtile[wave * 16][32], DH, lmod, lhalf);

  float mrow[8], lrow[8];
  v8f acc[4] = {};
#pragma unroll
  for (int v = 0; v < 8; v++) { mrow[v] = -1e30f; lrow[v] = 0.f; }

  const int nkt = (q0 + 128) / 32;
  for (int kt = 0; kt < nkt; kt++) {
    const int j0 = kt * 32;
    __syncthreads();
    for (int i = tid * 8; i < 32 * DH; i += 2048)
      *(h8*)(&ktile[0][0] + i) = *(const h8*)(Kp + (size_t)j0 * DH + i);
    {  // V transposed: one vectorized global load, 8 scattered LDS stores
      int r = (tid * 8) >> 6, cc = (tid * 8) & 63;
      h8 vv = *(const h8*)(Vp + (size_t)(j0 + r) * DH + cc);
#pragma unroll
      for (int e = 0; e < 8; e++) vtile[cc + e][r] = vv[e];
    }
    __syncthreads();

    if (j0 <= qrow + 15) {                    // wave-uniform: EXEC stays full
      v8f s[2];
#pragma unroll
      for (int t = 0; t < 2; t++) {
        int jl = t * 16 + lmod;
        v16h b0 = *(const v16h*)&ktile[jl][lhalf * 16];
        v16h b1 = *(const v16h*)&ktile[jl][32 + lhalf * 16];
        v8f sc = {};
        sc = wmma_f16(aq0, b0, sc);
        sc = wmma_f16(aq1, b1, sc);
        s[t] = sc;
      }
#pragma unroll
      for (int v = 0; v < 8; v++) {
        int m = lhalf * 8 + v;
        int qi = qrow + m;
        float s0 = (j0 + lmod <= qi) ? s[0][v] : -1e30f;
        float s1 = (j0 + 16 + lmod <= qi) ? s[1][v] : -1e30f;
        float mx = fmaxf(s0, s1);
#pragma unroll
        for (int off = 8; off >= 1; off >>= 1)
          mx = fmaxf(mx, __shfl_xor(mx, off, 16));
        float mnew = fmaxf(mrow[v], mx);
        float scale = __expf(mrow[v] - mnew);
        float p0 = __expf(s0 - mnew), p1 = __expf(s1 - mnew);
        float ps = p0 + p1;
#pragma unroll
        for (int off = 8; off >= 1; off >>= 1) ps += __shfl_xor(ps, off, 16);
        lrow[v] = lrow[v] * scale + ps;
        mrow[v] = mnew;
#pragma unroll
        for (int vt = 0; vt < 4; vt++) acc[vt][v] *= scale;
        pbuf[wave][m][lmod] = (_Float16)p0;
        pbuf[wave][m][16 + lmod] = (_Float16)p1;
      }
      // Same-wave LDS ops are in-order: safe to read pbuf back now.
      v16h ap = load_afrag(&pbuf[wave][0][0], 32, lmod, lhalf);
#pragma unroll
      for (int vt = 0; vt < 4; vt++) {
        v16h bv = *(const v16h*)&vtile[vt * 16 + lmod][lhalf * 16];
        acc[vt] = wmma_f16(ap, bv, acc[vt]);
      }
    }
  }

  const float hs = head_scale[h];
#pragma unroll
  for (int v = 0; v < 8; v++) {
    int m = lhalf * 8 + v;
    int qi = qrow + m;
    float inv = hs / lrow[v];
#pragma unroll
    for (int vt = 0; vt < 4; vt++) {
      int d = vt * 16 + lmod;
      AO[((size_t)bi * SEQ + qi) * DIM + h * DH + d] = (_Float16)(acc[vt][v] * inv);
    }
  }
}

// ---------------------------------------------------------------------------
// Kernel 3: out = ls_scale * LayerNorm(attn @ Wo^T + bo), written (n,b,512) f32
// ---------------------------------------------------------------------------
__global__ __launch_bounds__(256) void proj_ln_kernel(
    const _Float16* __restrict__ A_in, const _Float16* __restrict__ Wo16,
    const float* __restrict__ bo, const float* __restrict__ g,
    const float* __restrict__ bln, const float* __restrict__ ls,
    float* __restrict__ out) {
  __shared__ __align__(32) _Float16 lx[16][DIM];
  __shared__ float res[16][DIM];
  __shared__ float smu[16], srstd[16];
  const int n0 = blockIdx.x * 16, bi = blockIdx.y;
  const int tid = threadIdx.x, wave = tid >> 5, lane = tid & 31;
  const int lhalf = lane >> 4, lmod = lane & 15;

  for (int i = tid * 8; i < 16 * DIM; i += 2048)
    *(h8*)(&lx[0][0] + i) =
        *(const h8*)(A_in + ((size_t)bi * SEQ + n0) * DIM + i);
  __syncthreads();

  for (int t = wave; t < 32; t += 8) {
    const int oc0 = t * 16;
    v8f c = {};
#pragma unroll 4
    for (int kk = 0; kk < DIM; kk += 32) {
      v16h a = load_afrag(&lx[0][kk], DIM, lmod, lhalf);
      v16h bf =
          *(const v16h*)(Wo16 + (size_t)(oc0 + lmod) * DIM + kk + lhalf * 16);
      c = wmma_f16(a, bf, c);
    }
    const int o = oc0 + lmod;
#pragma unroll
    for (int v = 0; v < 8; v++) res[lhalf * 8 + v][o] = c[v] + bo[o];
  }
  __syncthreads();

  {  // LayerNorm stats: wave w -> rows 2w (lanes 0-15) and 2w+1 (lanes 16-31)
    int r = 2 * wave + lhalf;
    float s = 0.f, ss = 0.f;
    for (int c0 = lmod; c0 < DIM; c0 += 16) {
      float xv = res[r][c0];
      s += xv; ss += xv * xv;
    }
#pragma unroll
    for (int off = 8; off >= 1; off >>= 1) {
      s += __shfl_xor(s, off, 16);
      ss += __shfl_xor(ss, off, 16);
    }
    float mu = s * (1.f / DIM);
    float var = ss * (1.f / DIM) - mu * mu;
    if (lmod == 0) { smu[r] = mu; srstd[r] = rsqrtf(var + 1e-5f); }
  }
  __syncthreads();

  for (int i = tid; i < 16 * DIM; i += 256) {
    int r = i >> 9, c = i & 511;
    float y = (res[r][c] - smu[r]) * srstd[r] * g[c] + bln[c];
    out[((size_t)(n0 + r) * BATCH + bi) * DIM + c] = ls[c] * y;
  }
}

// ---------------------------------------------------------------------------
extern "C" void kernel_launch(void* const* d_in, const int* in_sizes, int n_in,
                              void* d_out, int out_size, void* d_ws,
                              size_t ws_size, hipStream_t stream) {
  const float* x    = (const float*)d_in[0];
  const float* rope = (const float*)d_in[1];
  const float* Wq   = (const float*)d_in[2];
  const float* Wk   = (const float*)d_in[3];
  const float* Wv   = (const float*)d_in[4];
  const float* Wo   = (const float*)d_in[5];
  const float* bo   = (const float*)d_in[6];
  const float* ln_g = (const float*)d_in[7];
  const float* ln_b = (const float*)d_in[8];
  const float* hs   = (const float*)d_in[9];
  const float* ls   = (const float*)d_in[10];
  float* out = (float*)d_out;

  const size_t qkv_elems = (size_t)BATCH * HEADS * SEQ * DH;  // 4.19M halves
  _Float16* Qb  = (_Float16*)d_ws;
  _Float16* Kb  = Qb + qkv_elems;
  _Float16* Vb  = Kb + qkv_elems;
  _Float16* Ab  = Vb + qkv_elems;                      // [b][n][512]
  _Float16* W16 = Ab + (size_t)BATCH * SEQ * DIM;      // 4 x 512x512 f16

  dim3 blk(256);
  wcvt_kernel<<<dim3(256, 4), blk, 0, stream>>>(Wq, Wk, Wv, Wo, W16);
  qkv_rope_kernel<<<dim3(SEQ / 16, BATCH), blk, 0, stream>>>(
      x, rope, W16, Qb, Kb, Vb);
  attn_kernel<<<dim3(SEQ / 128, BATCH * HEADS), blk, 0, stream>>>(
      Qb, Kb, Vb, hs, Ab);
  proj_ln_kernel<<<dim3(SEQ / 16, BATCH), blk, 0, stream>>>(
      Ab, W16 + (size_t)3 * DIM * DIM, bo, ln_g, ln_b, ls, out);
}
